// CopyLayer_39367670235353
// MI455X (gfx1250) — matrix operand order repaired and checked by reading, not compiled, not run on hardware
//
#include <hip/hip_runtime.h>

// Problem sizes (compile-time constants from the reference)
#define NB   4
#define LB   512
#define SB   512
#define DD   1024
#define VV   32000
#define ROWS (NB * LB)       // 2048

#define BM 128
#define BN 128
#define BK 32
#define KT (DD / BK)         // 32 k-steps

typedef __attribute__((ext_vector_type(16))) __bf16 v16bf;
typedef __attribute__((ext_vector_type(8)))  float  v8f;

union FragAB { v16bf v; uint4 u[2]; };
union FragC  { v8f   v; float f[8]; };

static __device__ __forceinline__ unsigned short f2bf(float f) {
    unsigned u = __float_as_uint(f);
    u += 0x7fffu + ((u >> 16) & 1u);          // round-to-nearest-even
    return (unsigned short)(u >> 16);
}

// CDNA5 async global->LDS copy (ASYNCcnt-tracked), 16 bytes per lane.
// vdst = per-lane LDS byte address, vaddr = per-lane 64-bit global address.
static __device__ __forceinline__ void async_b128(unsigned lds, const void* g) {
    asm volatile("global_load_async_to_lds_b128 %0, %1, off"
                 :: "v"(lds), "v"((unsigned long long)(uintptr_t)g)
                 : "memory");
}

// ---------------------------------------------------------------------------
// Kernel 1: gate = sigmoid(X·w_copy + b_copy), and X (f32) -> bf16 copy
// ---------------------------------------------------------------------------
__global__ __launch_bounds__(256)
void prep_kernel(const float* __restrict__ ds, const float* __restrict__ wc,
                 const float* __restrict__ bc, unsigned short* __restrict__ aBf,
                 float* __restrict__ gate) {
    __shared__ float red[256];
    const int row = blockIdx.x;
    const int d   = threadIdx.x * 4;
    const float4 x = *reinterpret_cast<const float4*>(ds + (size_t)row * DD + d);
    const float4 w = *reinterpret_cast<const float4*>(wc + d);
    ushort4 h;
    h.x = f2bf(x.x); h.y = f2bf(x.y); h.z = f2bf(x.z); h.w = f2bf(x.w);
    *reinterpret_cast<ushort4*>(aBf + (size_t)row * DD + d) = h;
    red[threadIdx.x] = x.x * w.x + x.y * w.y + x.z * w.z + x.w * w.w;
    __syncthreads();
    for (int off = 128; off > 0; off >>= 1) {
        if (threadIdx.x < off) red[threadIdx.x] += red[threadIdx.x + off];
        __syncthreads();
    }
    if (threadIdx.x == 0)
        gate[row] = 1.f / (1.f + __expf(-(red[0] + bc[0])));
}

// ---------------------------------------------------------------------------
// Kernel 2: w_gen (D x V, f32) -> wT (V x D, bf16) tiled transpose via LDS
// ---------------------------------------------------------------------------
__global__ __launch_bounds__(256)
void transpose_kernel(const float* __restrict__ wg, unsigned short* __restrict__ wT) {
    __shared__ float t[32][33];
    const int v0 = blockIdx.x * 32, d0 = blockIdx.y * 32;
    const int tx = threadIdx.x, ty = threadIdx.y;
#pragma unroll
    for (int i = 0; i < 4; ++i) {
        const int d = d0 + ty + i * 8;
        t[ty + i * 8][tx] = wg[(size_t)d * VV + v0 + tx];
    }
    __syncthreads();
#pragma unroll
    for (int i = 0; i < 4; ++i) {
        const int v = v0 + ty + i * 8;
        wT[(size_t)v * DD + d0 + tx] = f2bf(t[tx][ty + i * 8]);
    }
}

// ---------------------------------------------------------------------------
// Kernel 3: logits = Xbf16 · Wgen_bf16 + b_gen  (WMMA bf16, f32 accum)
// Triple-buffered async global->LDS staging (ASYNCcnt), 8 waves per WG,
// each wave a 32x64 tile = 8 WMMA accumulators; bias fused in epilogue.
// ---------------------------------------------------------------------------
__global__ __launch_bounds__(256)
void gemm_kernel(const unsigned short* __restrict__ A,
                 const unsigned short* __restrict__ Bt,
                 const float* __restrict__ bias,
                 float* __restrict__ out) {
    __shared__ __align__(16) unsigned short As[3][BM * BK];
    __shared__ __align__(16) unsigned short Bs[3][BN * BK];

    const int tid  = threadIdx.x;
    const int lane = tid & 31;
    const int wid  = tid >> 5;
    const int waveM = (wid >> 1) * 32;   // 0,32,64,96
    const int waveN = (wid & 1) * 64;    // 0,64
    const int mTile = blockIdx.y * BM;
    const int nTile = blockIdx.x * BN;

    // staging: thread t covers tile row t/2, 16 bf16 (2 async b128 per matrix)
    const int sRow = tid >> 1;
    const int sK   = (tid & 1) * 16;
    const unsigned short* gAp = A  + (size_t)(mTile + sRow) * DD + sK;
    const unsigned short* gBp = Bt + (size_t)(nTile + sRow) * DD + sK;

    unsigned ldsA[3], ldsB[3];
#pragma unroll
    for (int b = 0; b < 3; ++b) {
        ldsA[b] = (unsigned)(uintptr_t)&As[b][sRow * BK + sK];
        ldsB[b] = (unsigned)(uintptr_t)&Bs[b][sRow * BK + sK];
    }

    FragC c[2][4];
#pragma unroll
    for (int i = 0; i < 2; ++i)
#pragma unroll
        for (int j = 0; j < 4; ++j)
#pragma unroll
            for (int k = 0; k < 8; ++k) c[i][j].f[k] = 0.f;

    // prologue: stage k-step 0 into buffer 0
    async_b128(ldsA[0],      gAp);
    async_b128(ldsA[0] + 16, gAp + 8);
    async_b128(ldsB[0],      gBp);
    async_b128(ldsB[0] + 16, gBp + 8);

    const int mrow   = lane & 15;
    const int khalf  = (lane >> 4) * 8;    // unused; A and B both index by kstart
    const int kstart = (lane >> 4) * 16;   // A and B: K range per lane half

    int bufCur = 0;
    for (int kt = 0; kt < KT; ++kt) {
        const int bufNext = (bufCur == 2) ? 0 : bufCur + 1;
        if (kt + 1 < KT) {
            const unsigned short* a1 = gAp + (kt + 1) * BK;
            const unsigned short* b1 = gBp + (kt + 1) * BK;
            async_b128(ldsA[bufNext],      a1);
            async_b128(ldsA[bufNext] + 16, a1 + 8);
            async_b128(ldsB[bufNext],      b1);
            async_b128(ldsB[bufNext] + 16, b1 + 8);
            if (kt + 2 < KT) {             // pull k-step kt+2 into L2
                __builtin_prefetch(gAp + (kt + 2) * BK, 0, 0);
                __builtin_prefetch(gBp + (kt + 2) * BK, 0, 0);
            }
            asm volatile("s_wait_asynccnt 0x4" ::: "memory"); // group kt landed
        } else {
            asm volatile("s_wait_asynccnt 0x0" ::: "memory");
        }
        __syncthreads();   // publish group kt to all waves; fences buf reuse

        // load fragments (contiguous K: lane half h holds k = 16h..16h+15, same for A and B)
        FragAB a[2], b[4];
#pragma unroll
        for (int fm = 0; fm < 2; ++fm) {
            const uint4* p = reinterpret_cast<const uint4*>(
                &As[bufCur][(waveM + fm * 16 + mrow) * BK + kstart]);
            a[fm].u[0] = p[0];       // K: kstart .. kstart+7
            a[fm].u[1] = p[1];       // K: kstart+8 .. kstart+15
        }
#pragma unroll
        for (int fn = 0; fn < 4; ++fn) {
            const uint4* q = reinterpret_cast<const uint4*>(
                &Bs[bufCur][(waveN + fn * 16 + mrow) * BK + kstart]);
            b[fn].u[0] = q[0];       // K: kstart .. kstart+7
            b[fn].u[1] = q[1];       // K: kstart+8 .. kstart+15
        }

#pragma unroll
        for (int fm = 0; fm < 2; ++fm)
#pragma unroll
            for (int fn = 0; fn < 4; ++fn)
                c[fm][fn].v = __builtin_amdgcn_wmma_f32_16x16x32_bf16(
                    /*neg_a=*/false, a[fm].v, /*neg_b=*/false, b[fn].v,
                    /*c_mod=*/(short)0, c[fm][fn].v,
                    /*reuse_a=*/false, /*reuse_b=*/false);

        bufCur = bufNext;
    }

    // epilogue: C/D layout — m = v + 8*(lane>=16), n = lane&15; fuse b_gen
#pragma unroll
    for (int fn = 0; fn < 4; ++fn) {
        const int gn = nTile + waveN + fn * 16 + (lane & 15);
        const float bb = bias[gn];
#pragma unroll
        for (int fm = 0; fm < 2; ++fm) {
            const int gmBase = mTile + waveM + fm * 16 + 8 * (lane >> 4);
#pragma unroll
            for (int v = 0; v < 8; ++v)
                out[(size_t)(gmBase + v) * VV + gn] = c[fm][fn].f[v] + bb;
        }
    }
}

// ---------------------------------------------------------------------------
// Kernel 4: in-place  out = gate * softmax(out)  per row (online softmax)
// ---------------------------------------------------------------------------
__global__ __launch_bounds__(256)
void softmax_kernel(float* __restrict__ out, const float* __restrict__ gate) {
    __shared__ float rm[256], rs[256];
    const int row = blockIdx.x;
    float* p = out + (size_t)row * VV;
    float m = -3.4e38f, s = 0.f;
    for (int v = threadIdx.x; v < VV; v += 256) {
        const float x = p[v];
        if (x > m) { s = s * __expf(m - x) + 1.f; m = x; }
        else       { s += __expf(x - m); }
    }
    rm[threadIdx.x] = m; rs[threadIdx.x] = s;
    __syncthreads();
    for (int off = 128; off > 0; off >>= 1) {
        if (threadIdx.x < off) {
            const float m1 = rm[threadIdx.x], s1 = rs[threadIdx.x];
            const float m2 = rm[threadIdx.x + off], s2 = rs[threadIdx.x + off];
            const float M = fmaxf(m1, m2);
            rs[threadIdx.x] = s1 * __expf(m1 - M) + s2 * __expf(m2 - M);
            rm[threadIdx.x] = M;
        }
        __syncthreads();
    }
    const float M = rm[0];
    const float inv = gate[row] / rs[0];
    for (int v = threadIdx.x; v < VV; v += 256)
        p[v] = __expf(p[v] - M) * inv;
}

// ---------------------------------------------------------------------------
// Kernel 5: out[row, ids[n,s]] += (1-gate[row]) * attn[row, s]
// ---------------------------------------------------------------------------
__global__ __launch_bounds__(512)
void scatter_kernel(const float* __restrict__ attn, const int* __restrict__ ids,
                    const float* __restrict__ gate, float* __restrict__ out) {
    const int row = blockIdx.x;          // n*L + l
    const int n   = row >> 9;            // L = 512
    const int s   = threadIdx.x;
    const int id  = ids[n * SB + s];
    const float g = 1.f - gate[row];
    atomicAdd(out + (size_t)row * VV + id, g * attn[(size_t)row * SB + s]);
}

// ---------------------------------------------------------------------------
extern "C" void kernel_launch(void* const* d_in, const int* in_sizes, int n_in,
                              void* d_out, int out_size, void* d_ws, size_t ws_size,
                              hipStream_t stream) {
    (void)in_sizes; (void)n_in; (void)out_size; (void)ws_size;
    const float* ds    = (const float*)d_in[0];   // (N,L,D)
    const float* attn  = (const float*)d_in[1];   // (N,L,S)
    const int*   ids   = (const int*)  d_in[2];   // (N,S)
    const float* wcopy = (const float*)d_in[3];   // (D,1)
    const float* bcopy = (const float*)d_in[4];   // (1,)
    const float* wgen  = (const float*)d_in[5];   // (D,V)
    const float* bgen  = (const float*)d_in[6];   // (V,)
    float* out = (float*)d_out;

    // workspace layout
    float*          gate = (float*)d_ws;                                    // 8 KB
    unsigned short* aBf  = (unsigned short*)((char*)d_ws + 8192);           // 4 MB
    unsigned short* wT   = (unsigned short*)((char*)d_ws + 8192
                                             + (size_t)ROWS * DD * 2);      // 64 MB

    prep_kernel<<<ROWS, 256, 0, stream>>>(ds, wcopy, bcopy, aBf, gate);
    transpose_kernel<<<dim3(VV / 32, DD / 32), dim3(32, 8), 0, stream>>>(wgen, wT);
    gemm_kernel<<<dim3(VV / BN, ROWS / BM), 256, 0, stream>>>(aBf, wT, bgen, out);
    softmax_kernel<<<ROWS, 256, 0, stream>>>(out, gate);
    scatter_kernel<<<ROWS, 512, 0, stream>>>(attn, ids, gate, out);
}
